// MultiQueryAttention_59588376264816
// MI455X (gfx1250) — matrix operand order repaired and checked by reading, not compiled, not run on hardware
//
#include <hip/hip_runtime.h>
#include <hip/hip_bf16.h>

typedef __attribute__((ext_vector_type(16))) _Float16 v16h;
typedef __attribute__((ext_vector_type(8)))  _Float16 v8h;
typedef __attribute__((ext_vector_type(8)))  float    v8f;

#define NE    2048   // n_embd
#define NH    16     // heads
#define HD    128    // head dim
#define BB    2      // batch
#define SS    2048   // seq len
#define MROWS (BB*SS)

// ---------------------------------------------------------------- f32 -> f16
__global__ void cvt_f32_f16(const float* __restrict__ in, _Float16* __restrict__ out, int n) {
    int i = blockIdx.x * blockDim.x + threadIdx.x;
    if (i < n) out[i] = (_Float16)in[i];
}

// ----------------------------------------- V [B,S,HD] -> Vt [B,HD,S] (f16)
__global__ void transpose_v(const _Float16* __restrict__ Vh, _Float16* __restrict__ Vt) {
    int i = blockIdx.x * blockDim.x + threadIdx.x;     // over BB*SS*HD
    if (i >= BB * SS * HD) return;
    const int d = i & (HD - 1);
    const int s = (i >> 7) & (SS - 1);
    const int b = i >> 18;
    Vt[((size_t)b * HD + d) * SS + s] = Vh[i];
}

// ---------------------- C[M,N] = A[M,K] @ Bw[N,K]^T + bias ; 16x64 strip/wave
template<bool OUT_F16>
__global__ __launch_bounds__(32)
void gemm_bt16(const _Float16* __restrict__ A,
               const _Float16* __restrict__ Bw,
               const float*    __restrict__ bias,
               void*           __restrict__ Cout,
               int N, int K) {
    const int lane = threadIdx.x & 31;
    const int lid  = lane & 15;
    const int half = lane >> 4;
    const int n0   = blockIdx.x * 64;
    const int m0   = blockIdx.y * 16;

    const _Float16* arow = A + (size_t)(m0 + lid) * K;      // A-frag: lane = row
    const _Float16* brow[4];
#pragma unroll
    for (int t = 0; t < 4; ++t)                              // B-frag: lane = col
        brow[t] = Bw + (size_t)(n0 + t * 16 + lid) * K;

    const v8f zero = {};
    v8f acc[4];
#pragma unroll
    for (int t = 0; t < 4; ++t) acc[t] = zero;

    for (int k0 = 0; k0 < K; k0 += 32) {
        // A frag in this code's contiguous k order (k = 16*half + i), same as B
        v8h alo = *(const v8h*)(arow + k0 + 16 * half);
        v8h ahi = *(const v8h*)(arow + k0 + 16 * half + 8);
        v16h a;
#pragma unroll
        for (int i = 0; i < 8; ++i) { a[i] = alo[i]; a[8 + i] = ahi[i]; }
#pragma unroll
        for (int t = 0; t < 4; ++t) {
            // B frag: 16 contiguous K halves per lane, same k order as A
            v16h b = *(const v16h*)(brow[t] + k0 + 16 * half);
            acc[t] = __builtin_amdgcn_wmma_f32_16x16x32_f16(false, a, false, b,
                                                            (short)0, acc[t], false, false);
        }
    }
#pragma unroll
    for (int t = 0; t < 4; ++t) {
        const float bn = bias[n0 + t * 16 + lid];
#pragma unroll
        for (int v = 0; v < 8; ++v) {           // C layout: row = v + 8*half, col = lid
            const int row = m0 + v + 8 * half;
            const float val = acc[t][v] + bn;
            if (OUT_F16) ((_Float16*)Cout)[(size_t)row * N + n0 + t * 16 + lid] = (_Float16)val;
            else         ((float*)   Cout)[(size_t)row * N + n0 + t * 16 + lid] = val;
        }
    }
}

// ---------------------------------------------------- flash multi-query attention
// One wave per (b, h, 16-query tile). K/V shared across heads (MQA).
__global__ __launch_bounds__(32)
void mqa_flash(const _Float16* __restrict__ Q,   // [B, S, NH*HD]
               const _Float16* __restrict__ Kc,  // [B, S, HD]
               const _Float16* __restrict__ Vt,  // [B, HD, S]  (transposed)
               _Float16*       __restrict__ Oa)  // [B, S, NH*HD]
{
    const int lane = threadIdx.x & 31;
    const int lid  = lane & 15;
    const int half = lane >> 4;
    const int qt   = blockIdx.x;   // query tile (16 rows)
    const int h    = blockIdx.y;
    const int b    = blockIdx.z;
    const float scale = 0.08838834764831845f;   // 1/sqrt(128)

    __shared__ _Float16 Plds[16 * 32];          // P tile re-layout buffer (1 KB)

    // ---- load Q tile as 4 A-fragments covering d = 0..127 (kept in VGPRs)
    const _Float16* qrow =
        Q + ((size_t)(b * SS + qt * 16 + lid)) * (size_t)(NH * HD) + (size_t)h * HD;
    v16h qf[4];
#pragma unroll
    for (int c = 0; c < 4; ++c) {
        v8h lo = *(const v8h*)(qrow + c * 32 + 16 * half);
        v8h hi = *(const v8h*)(qrow + c * 32 + 16 * half + 8);
#pragma unroll
        for (int i = 0; i < 8; ++i) { qf[c][i] = lo[i]; qf[c][8 + i] = hi[i]; }
    }

    const v8f zero = {};
    v8f o[8];                                    // 16q x 128d accumulator (8 C-frags)
#pragma unroll
    for (int c = 0; c < 8; ++c) o[c] = zero;
    float mrow[8], lrow[8];                      // per-owned-row online-softmax stats
#pragma unroll
    for (int v = 0; v < 8; ++v) { mrow[v] = -1e30f; lrow[v] = 0.0f; }

    const int lastkey = qt * 16 + 15;
    const int nkt     = lastkey / 32 + 1;        // causal: skip fully-masked key tiles

    const _Float16* vhead = Vt + (size_t)b * HD * SS;

    for (int j = 0; j < nkt; ++j) {
        const int kb = j * 32;

        // ---- scores: two 16x16 fragments over a 32-key tile, 8 WMMAs
        v8f s_lo = zero, s_hi = zero;
        const _Float16* krlo = Kc + ((size_t)(b * SS + kb + lid))      * HD;
        const _Float16* krhi = Kc + ((size_t)(b * SS + kb + 16 + lid)) * HD;
#pragma unroll
        for (int c = 0; c < 4; ++c) {
            v16h bk0 = *(const v16h*)(krlo + c * 32 + 16 * half);
            s_lo = __builtin_amdgcn_wmma_f32_16x16x32_f16(false, qf[c], false, bk0,
                                                          (short)0, s_lo, false, false);
            v16h bk1 = *(const v16h*)(krhi + c * 32 + 16 * half);
            s_hi = __builtin_amdgcn_wmma_f32_16x16x32_f16(false, qf[c], false, bk1,
                                                          (short)0, s_hi, false, false);
        }

        // ---- online softmax (row-wise across the 16-lane half-groups)
#pragma unroll
        for (int v = 0; v < 8; ++v) {
            const int rowg = qt * 16 + v + 8 * half;
            float sl = s_lo[v] * scale;
            float sh = s_hi[v] * scale;
            if (kb + lid      > rowg) sl = -1e30f;   // causal mask
            if (kb + 16 + lid > rowg) sh = -1e30f;
            float mx = fmaxf(sl, sh);
#pragma unroll
            for (int off = 1; off < 16; off <<= 1) mx = fmaxf(mx, __shfl_xor(mx, off, 32));
            const float mold = mrow[v];
            const float mn   = fmaxf(mold, mx);
            const float pl   = __expf(sl - mn);
            const float ph   = __expf(sh - mn);
            float rs = pl + ph;
#pragma unroll
            for (int off = 1; off < 16; off <<= 1) rs += __shfl_xor(rs, off, 32);
            const float alpha = __expf(mold - mn);
            mrow[v] = mn;
            lrow[v] = lrow[v] * alpha + rs;
#pragma unroll
            for (int c = 0; c < 8; ++c) o[c][v] *= alpha;
            const int prow = v + 8 * half;
            Plds[prow * 32 + lid]      = (_Float16)pl;   // C-layout -> LDS
            Plds[prow * 32 + 16 + lid] = (_Float16)ph;
        }
        __syncthreads();

        // ---- reload P in A-fragment layout (16 rows x 32 keys)
        v16h pa;
        {
            const _Float16* pr = &Plds[lid * 32];
            v8h lo = *(const v8h*)(pr + 16 * half);
            v8h hi = *(const v8h*)(pr + 16 * half + 8);
#pragma unroll
            for (int i = 0; i < 8; ++i) { pa[i] = lo[i]; pa[8 + i] = hi[i]; }
        }

        // ---- O += P @ V : 8 WMMAs; B-frags are contiguous v16h loads from Vt
#pragma unroll
        for (int c = 0; c < 8; ++c) {
            const _Float16* vcol = vhead + (size_t)(c * 16 + lid) * SS + kb + 16 * half;
            v16h bv = *(const v16h*)vcol;
            o[c] = __builtin_amdgcn_wmma_f32_16x16x32_f16(false, pa, false, bv,
                                                          (short)0, o[c], false, false);
        }
        __syncthreads();
    }

    // ---- normalize and store attention output tile (f16, [B,S,NH*HD])
    _Float16* obase = Oa + ((size_t)(b * SS + qt * 16)) * (NH * HD) + (size_t)h * HD;
#pragma unroll
    for (int v = 0; v < 8; ++v) {
        const float inv = 1.0f / lrow[v];
        const int row = v + 8 * half;
#pragma unroll
        for (int c = 0; c < 8; ++c)
            obase[(size_t)row * (NH * HD) + c * 16 + lid] = (_Float16)(o[c][v] * inv);
    }
}

// ---------------------------------------------------------------------- launch
extern "C" void kernel_launch(void* const* d_in, const int* in_sizes, int n_in,
                              void* d_out, int out_size, void* d_ws, size_t ws_size,
                              hipStream_t stream) {
    const float* x  = (const float*)d_in[0];
    const float* Wq = (const float*)d_in[1];
    const float* bq = (const float*)d_in[2];
    const float* Wk = (const float*)d_in[3];
    const float* bk = (const float*)d_in[4];
    const float* Wv = (const float*)d_in[5];
    const float* bv = (const float*)d_in[6];
    const float* Wo = (const float*)d_in[7];
    const float* bo = (const float*)d_in[8];
    float* out = (float*)d_out;

    char*  ws  = (char*)d_ws;
    size_t off = 0;
    auto alloc = [&](size_t bytes) -> void* {
        void* p = ws + off;
        off += (bytes + 255) & ~(size_t)255;
        return p;
    };
    _Float16* Xh  = (_Float16*)alloc((size_t)MROWS * NE * 2);
    _Float16* Wqh = (_Float16*)alloc((size_t)NE * NE * 2);
    _Float16* Wkh = (_Float16*)alloc((size_t)HD * NE * 2);
    _Float16* Wvh = (_Float16*)alloc((size_t)HD * NE * 2);
    _Float16* Woh = (_Float16*)alloc((size_t)NE * NE * 2);
    _Float16* Qh  = (_Float16*)alloc((size_t)MROWS * NE * 2);
    _Float16* Kh  = (_Float16*)alloc((size_t)MROWS * HD * 2);
    _Float16* Vh  = (_Float16*)alloc((size_t)MROWS * HD * 2);
    _Float16* Vtr = (_Float16*)alloc((size_t)MROWS * HD * 2);
    _Float16* Ah  = (_Float16*)alloc((size_t)MROWS * NE * 2);

    auto cvt = [&](const float* src, _Float16* dst, int n) {
        cvt_f32_f16<<<(n + 255) / 256, 256, 0, stream>>>(src, dst, n);
    };
    cvt(x,  Xh,  MROWS * NE);
    cvt(Wq, Wqh, NE * NE);
    cvt(Wk, Wkh, HD * NE);
    cvt(Wv, Wvh, HD * NE);
    cvt(Wo, Woh, NE * NE);

    // Q/K/V projections
    gemm_bt16<true><<<dim3(NE / 64, MROWS / 16), 32, 0, stream>>>(Xh, Wqh, bq, Qh, NE, NE);
    gemm_bt16<true><<<dim3(HD / 64, MROWS / 16), 32, 0, stream>>>(Xh, Wkh, bk, Kh, HD, NE);
    gemm_bt16<true><<<dim3(HD / 64, MROWS / 16), 32, 0, stream>>>(Xh, Wvh, bv, Vh, HD, NE);

    // V -> V^T so attention B-fragments are contiguous loads
    transpose_v<<<(MROWS * HD + 255) / 256, 256, 0, stream>>>(Vh, Vtr);

    // fused causal multi-query attention
    mqa_flash<<<dim3(SS / 16, NH, BB), 32, 0, stream>>>(Qh, Kh, Vtr, Ah);

    // output projection (f32 result + bias straight to d_out)
    gemm_bt16<false><<<dim3(NE / 64, MROWS / 16), 32, 0, stream>>>(Ah, Woh, bo, out, NE, NE);
}
